// Model_54245436948893
// MI455X (gfx1250) — matrix operand order repaired and checked by reading, not compiled, not run on hardware
//
#include <hip/hip_runtime.h>
#include <hip/hip_bf16.h>
#include <math.h>

typedef __attribute__((ext_vector_type(16))) __bf16 v16bf;
typedef __attribute__((ext_vector_type(8)))  __bf16 v8bf;
typedef __attribute__((ext_vector_type(4)))  __bf16 v4bf;
typedef __attribute__((ext_vector_type(2)))  __bf16 v2bf;
typedef __attribute__((ext_vector_type(8)))  float  v8f;
typedef __attribute__((ext_vector_type(4)))  float  v4f;
typedef __attribute__((ext_vector_type(2)))  float  v2f;

#define B_   64
#define T_   4096
#define HID_ 128
#define G4   512   // 4*HID
#define DIN  22

#define LOG2E_F 1.4426950408889634f

// Branchless fast sigmoid/tanh: v_exp_f32 + v_rcp_f32, correct saturation at +-inf.
__device__ __forceinline__ float fast_sig(float x) {
  float e = __builtin_amdgcn_exp2f(-LOG2E_F * x);
  return __builtin_amdgcn_rcpf(1.0f + e);
}
__device__ __forceinline__ float fast_tanh(float x) {
  float e = __builtin_amdgcn_exp2f(2.0f * LOG2E_F * x);
  return 1.0f - 2.0f * __builtin_amdgcn_rcpf(e + 1.0f);
}

// Workgroup barrier that drains ONLY DScnt (LDS visibility). Unlike
// __syncthreads(), it does NOT wait loadcnt, so in-flight non-temporal
// global prefetches (to private VGPRs) stay in flight across the barrier.
__device__ __forceinline__ void wg_barrier_lds() {
  asm volatile(
      "s_wait_dscnt 0x0\n\t"
      "s_barrier_signal -1\n\t"
      "s_barrier_wait -1" ::: "memory");
}

// ---------------------------------------------------------------------------
// Kernel 1: vectorized ODE update -> feats [B*T, 22] (fp32)
// ---------------------------------------------------------------------------
__global__ void feats_kernel(const float* __restrict__ x, const float* __restrict__ u,
                             const float* dtp, const float* theta,
                             const float* H_e, const float* tau_e,
                             const float* H_i, const float* tau_i,
                             const float* lam1, const float* lam2,
                             const float* lam3, const float* lam4,
                             const float* C_f, const float* C_l,
                             const float* C_u, const float* C_b,
                             float* __restrict__ feats) {
  int idx = blockIdx.x * blockDim.x + threadIdx.x;  // b*T + t
  if (idx >= B_ * T_) return;
  const float dt = dtp[0];
  float th[4], cf[4], cl[4], cu[4], cb[4];
#pragma unroll
  for (int i = 0; i < 4; i++) { th[i]=theta[i]; cf[i]=C_f[i]; cl[i]=C_l[i]; cu[i]=C_u[i]; cb[i]=C_b[i]; }
  const float he = H_e[0], te = tau_e[0], hi = H_i[0], ti = tau_i[0];
  const float l1 = lam1[0], l2 = lam2[0], l3 = lam3[0], l4 = lam4[0];
  const float He_te = he / te;
  const float inv_te = 1.0f / te, inv_te2 = inv_te * inv_te;
  const float inv_ti = 1.0f / ti, inv_ti2 = inv_ti * inv_ti;

  float xv[11][2];
  const float* xp = x + (size_t)idx * 22;
#pragma unroll
  for (int ch = 0; ch < 11; ch++) { xv[ch][0] = xp[ch*2]; xv[ch][1] = xp[ch*2+1]; }
  float uu[2];
  uu[0] = u[(size_t)idx*2]; uu[1] = u[(size_t)idx*2+1];

  float s_xt[2], ut_th[2];
#pragma unroll
  for (int m = 0; m < 2; m++) {
    float xt = xv[0][0]*th[0*2+m] + xv[0][1]*th[1*2+m];
    ut_th[m] = uu[0]*th[0*2+m] + uu[1]*th[1*2+m];
    s_xt[m]  = fast_sig(xt);
  }
  float o[11][2];
#pragma unroll
  for (int m = 0; m < 2; m++) {
    o[0][m] = xv[0][m] + dt * (xv[5][m] - xv[6][m]);
    o[1][m] = xv[1][m] + dt * xv[4][m];
    o[2][m] = xv[2][m] + dt * xv[5][m];
    o[3][m] = xv[3][m] + dt * xv[6][m];
    float m4 = 0.f, m5 = 0.f, m8 = 0.f, mu = 0.f;
#pragma unroll
    for (int n = 0; n < 2; n++) {
      float Id = (m == n) ? 1.0f : 0.0f;
      m4 += s_xt[n] * (cf[m*2+n] + cl[m*2+n] + l1 * Id);
      m5 += s_xt[n] * (cb[m*2+n] + cl[m*2+n]);
      m8 += s_xt[n] * (cb[m*2+n] + cl[m*2+n] + l3 * Id);
      mu += ut_th[n] * cu[m*2+n];
    }
    o[4][m] = xv[4][m] + dt * (He_te * (m4 + mu) - 2.0f*xv[4][m]*inv_te - xv[1][m]*inv_te2);
    o[5][m] = xv[5][m] + dt * (He_te * (m5 + l2 * fast_sig(xv[3][m])) - 2.0f*xv[5][m]*inv_te - xv[2][m]*inv_te2);
    o[6][m] = xv[6][m] + dt * ((hi/ti) * l4 * fast_sig(xv[7][m]) - 2.0f*xv[6][m]*inv_ti - xv[3][m]*inv_ti2);
    o[7][m] = xv[7][m] + dt * xv[8][m];
    o[8][m] = xv[8][m] + dt * (He_te * m8 - 2.0f*xv[8][m]*inv_te - xv[7][m]*inv_te2);
    o[9][m]  = xv[9][m];
    o[10][m] = xv[10][m];
  }
  float* fo = feats + (size_t)idx * 22;
#pragma unroll
  for (int ch = 0; ch < 11; ch++) { fo[ch*2] = o[ch][0]; fo[ch*2+1] = o[ch][1]; }
}

// ---------------------------------------------------------------------------
// Kernel 2: input projection gates[t][b][512] = feats @ W_ih^T + b_ih + b_hh
// K=22 too skinny for WMMA; bound by the 512MB NT store -> VALU.
// ---------------------------------------------------------------------------
__global__ __launch_bounds__(256) void ingemm_kernel(const float* __restrict__ feats,
                                                     const float* __restrict__ W_ih,
                                                     const float* __restrict__ b_ih,
                                                     const float* __restrict__ b_hh,
                                                     float* __restrict__ gates) {
  __shared__ float Wl[G4 * DIN];
  __shared__ float bl[G4];
  int tid = threadIdx.x;
  for (int i = tid; i < G4 * DIN; i += 256) Wl[i] = W_ih[i];
  for (int i = tid; i < G4; i += 256) bl[i] = b_ih[i] + b_hh[i];
  __syncthreads();
  int row = blockIdx.x * 2 + (tid >> 7);  // b*T + t
  int g0 = (tid & 127) * 4;
  float f[DIN];
  const float* fp = feats + (size_t)row * DIN;
#pragma unroll
  for (int i = 0; i < DIN; i++) f[i] = fp[i];
  v4f acc;
#pragma unroll
  for (int e = 0; e < 4; e++) acc[e] = bl[g0 + e];
#pragma unroll
  for (int i = 0; i < DIN; i++) {
#pragma unroll
    for (int e = 0; e < 4; e++) acc[e] += f[i] * Wl[(g0 + e) * DIN + i];
  }
  int b = row / T_, t = row % T_;
  v4f* gp = (v4f*)(gates + ((size_t)t * B_ + b) * G4 + g0);
  __builtin_nontemporal_store(acc, gp);
}

// ---------------------------------------------------------------------------
// Kernel 3: pack W_hh^T (bf16) into the WMMA-B fragment layout: one
// contiguous 32B read per lane per fragment in the recurrent kernel.
// B[k][n] = W_hh[n][k]; lane L: n = 16*nt + L%16, K-half = L/16, slot s -> k.
// ---------------------------------------------------------------------------
__global__ void pack_whh_kernel(const float* __restrict__ W_hh, __bf16* __restrict__ wpack) {
  int p = blockIdx.x * blockDim.x + threadIdx.x;  // 32*4*32*16 = 65536
  if (p >= 32 * 4 * 32 * 16) return;
  int slot = p & 15;
  int lane = (p >> 4) & 31;
  int kk   = (p >> 9) & 3;
  int nt   = p >> 11;
  int n = nt * 16 + (lane & 15);
  int k = kk * 32 + (lane >> 4) * 8 + (slot & 7) + (slot >> 3) * 16;
  wpack[p] = (__bf16)W_hh[n * HID_ + k];
}

// ---------------------------------------------------------------------------
// Kernel 4: sequential LSTM. 4 WGs x 16 batch rows. 32 waves (1024 thr) per
// WG, ONE N-tile per wave -> only 4 loop-invariant B fragments (32 VGPRs)
// per wave, so everything stays register-resident (no rematerialized global
// reloads). gates_in software-pipelined with NT loads across the DS-only
// barriers; branchless transcendentals; output projection one (b,o) pair
// per wave via full-wave shfl_xor reduction.
// ---------------------------------------------------------------------------
__global__ __launch_bounds__(1024, 1) void lstm_kernel(const float* __restrict__ gates,
                                                       const __bf16* __restrict__ wpack,
                                                       const float* __restrict__ W_lin,
                                                       const float* __restrict__ b_lin,
                                                       float* __restrict__ out) {
  __shared__ __attribute__((aligned(16))) __bf16 hbuf[16 * HID_];
  __shared__ __attribute__((aligned(16))) float gbuf[16 * G4];
  __shared__ __attribute__((aligned(16))) float wlin[2 * HID_];
  __shared__ float blin[2];

  const int tid  = threadIdx.x;
  const int lane = tid & 31;
  const int wave = tid >> 5;           // 0..31 == N-tile index
  const int b0   = blockIdx.x * 16;

  for (int i = tid; i < 2 * HID_; i += 1024) wlin[i] = W_lin[i];
  if (tid < 2) blin[tid] = b_lin[tid];
  for (int i = tid; i < 16 * HID_; i += 1024) hbuf[i] = (__bf16)0.0f;

  // Loop-invariant B fragments: wave owns N-tile `wave`, 4 K-slices.
  const int nt = wave;
  v16bf Breg[4];
#pragma unroll
  for (int kk = 0; kk < 4; kk++) {
    Breg[kk] = *(const v16bf*)(wpack + (((size_t)(nt * 4 + kk)) * 32 + lane) * 16);
  }

  // gate-update mapping: thread owns 2 hidden units of one batch row
  const int gb = tid >> 6;             // batch row 0..15
  const int j0 = (tid & 63) * 2;       // hidden unit 0..126
  float creg[2] = {0.f, 0.f};

  // output-projection mapping: one (b,o) pair per wave, 4 elems per lane
  const int pbb = wave >> 1;
  const int poo = wave & 1;

  // C/D fragment mapping: lane = n + 16*(m/8), vgpr r = m%8
  const int nl = lane & 15;
  const int mb = (lane >> 4) * 8;
  const int arow = lane & 15;
  const int akb  = (lane >> 4) * 8;

  // prefetch gates_in fragment for t = 0
  v8f cin;
  {
    const float* gp = gates + ((size_t)(b0 + mb)) * G4 + nt * 16 + nl;
#pragma unroll
    for (int r = 0; r < 8; r++) cin[r] = __builtin_nontemporal_load(gp + (size_t)r * G4);
  }

  for (int t = 0; t < T_; ++t) {
    wg_barrier_lds();  // h_{t-1} visible in hbuf; gbuf free

    // Build A fragments (16x32 bf16 per K-slice) from row-major h in LDS.
    v16bf A[4];
#pragma unroll
    for (int kk = 0; kk < 4; kk++) {
      const __bf16* hr = &hbuf[arow * HID_ + kk * 32 + akb];
      v8bf lo = *(const v8bf*)hr;          // K = akb .. akb+7
      v8bf hi = *(const v8bf*)(hr + 16);   // K = akb+16 .. akb+23
      v16bf a;
#pragma unroll
      for (int s = 0; s < 8; s++) { a[s] = lo[s]; a[s + 8] = hi[s]; }
      A[kk] = a;
    }

    // Distributed output projection for step t-1 (reads hbuf, all waves).
    if (t > 0) {
      v4bf hv = *(const v4bf*)&hbuf[pbb * HID_ + lane * 4];
      v4f  wl = *(const v4f*)&wlin[poo * HID_ + lane * 4];
      float part = 0.f;
#pragma unroll
      for (int e = 0; e < 4; e++) part += (float)hv[e] * wl[e];
      part += __shfl_xor(part, 1, 32);
      part += __shfl_xor(part, 2, 32);
      part += __shfl_xor(part, 4, 32);
      part += __shfl_xor(part, 8, 32);
      part += __shfl_xor(part, 16, 32);
      if (lane == 0)
        out[(((size_t)(b0 + pbb)) * T_ + (t - 1)) * 2 + poo] = part + blin[poo];
    }

    v8f acc = cin;
#pragma unroll
    for (int kk = 0; kk < 4; kk++)
      acc = __builtin_amdgcn_wmma_f32_16x16x32_bf16(false, A[kk], false, Breg[kk],
                                                    (short)0, acc, false, false);

    // Prefetch gates_in for t+1: stays in flight across both barriers.
    if (t + 1 < T_) {
      const float* gp = gates + ((size_t)(t + 1) * B_ + b0 + mb) * G4 + nt * 16 + nl;
#pragma unroll
      for (int r = 0; r < 8; r++) cin[r] = __builtin_nontemporal_load(gp + (size_t)r * G4);
    }

#pragma unroll
    for (int r = 0; r < 8; r++) gbuf[(mb + r) * G4 + nt * 16 + nl] = acc[r];

    wg_barrier_lds();  // full pre-activation gate vector ready

    // c/h update: i | f | g | o chunks of the 512-wide vector, 2 units/thread
    const v2f iv = *(const v2f*)&gbuf[gb * G4 + j0];
    const v2f fv = *(const v2f*)&gbuf[gb * G4 + HID_ + j0];
    const v2f gv = *(const v2f*)&gbuf[gb * G4 + 2 * HID_ + j0];
    const v2f ov = *(const v2f*)&gbuf[gb * G4 + 3 * HID_ + j0];
    v2bf hv;
#pragma unroll
    for (int e = 0; e < 2; e++) {
      float cc = fast_sig(fv[e]) * creg[e] + fast_sig(iv[e]) * fast_tanh(gv[e]);
      creg[e] = cc;
      hv[e] = (__bf16)(fast_sig(ov[e]) * fast_tanh(cc));
    }
    *(v2bf*)&hbuf[gb * HID_ + j0] = hv;  // one ds_store_b32
  }

  wg_barrier_lds();
  {  // output projection for the last step
    v4bf hv = *(const v4bf*)&hbuf[pbb * HID_ + lane * 4];
    v4f  wl = *(const v4f*)&wlin[poo * HID_ + lane * 4];
    float part = 0.f;
#pragma unroll
    for (int e = 0; e < 4; e++) part += (float)hv[e] * wl[e];
    part += __shfl_xor(part, 1, 32);
    part += __shfl_xor(part, 2, 32);
    part += __shfl_xor(part, 4, 32);
    part += __shfl_xor(part, 8, 32);
    part += __shfl_xor(part, 16, 32);
    if (lane == 0)
      out[(((size_t)(b0 + pbb)) * T_ + (T_ - 1)) * 2 + poo] = part + blin[poo];
  }
}

// ---------------------------------------------------------------------------
extern "C" void kernel_launch(void* const* d_in, const int* in_sizes, int n_in,
                              void* d_out, int out_size, void* d_ws, size_t ws_size,
                              hipStream_t stream) {
  const float* x     = (const float*)d_in[0];
  const float* u     = (const float*)d_in[1];
  const float* dt    = (const float*)d_in[2];
  const float* theta = (const float*)d_in[3];
  const float* H_e   = (const float*)d_in[4];
  const float* tau_e = (const float*)d_in[5];
  const float* H_i   = (const float*)d_in[6];
  const float* tau_i = (const float*)d_in[7];
  const float* lam1  = (const float*)d_in[8];
  const float* lam2  = (const float*)d_in[9];
  const float* lam3  = (const float*)d_in[10];
  const float* lam4  = (const float*)d_in[11];
  const float* C_f   = (const float*)d_in[12];
  const float* C_l   = (const float*)d_in[13];
  const float* C_u   = (const float*)d_in[14];
  const float* C_b   = (const float*)d_in[15];
  const float* W_ih  = (const float*)d_in[16];
  const float* W_hh  = (const float*)d_in[17];
  const float* b_ih  = (const float*)d_in[18];
  const float* b_hh  = (const float*)d_in[19];
  const float* W_lin = (const float*)d_in[20];
  const float* b_lin = (const float*)d_in[21];

  char* ws = (char*)d_ws;
  const size_t gates_bytes = (size_t)T_ * B_ * G4 * sizeof(float);   // 512 MB
  const size_t feats_bytes = (size_t)B_ * T_ * DIN * sizeof(float);  // 23 MB
  float*  gates = (float*)ws;
  float*  feats = (float*)(ws + gates_bytes);
  __bf16* wpack = (__bf16*)(ws + gates_bytes + feats_bytes);         // 128 KB

  feats_kernel<<<(B_ * T_ + 255) / 256, 256, 0, stream>>>(
      x, u, dt, theta, H_e, tau_e, H_i, tau_i, lam1, lam2, lam3, lam4,
      C_f, C_l, C_u, C_b, feats);
  ingemm_kernel<<<B_ * T_ / 2, 256, 0, stream>>>(feats, W_ih, b_ih, b_hh, gates);
  pack_whh_kernel<<<65536 / 256, 256, 0, stream>>>(W_hh, wpack);
  lstm_kernel<<<4, 1024, 0, stream>>>(gates, wpack, W_lin, b_lin, (float*)d_out);
}
